// CQAttention_78804059947436
// MI455X (gfx1250) — matrix operand order repaired and checked by reading, not compiled, not run on hardware
//
#include <hip/hip_runtime.h>
#include <hip/hip_bf16.h>
#include <math.h>

#define BB   64
#define DD   128
#define LCc  1024
#define LQq  128
#define NEGV (-1e30f)

typedef __attribute__((ext_vector_type(16))) __bf16 v16bf;
typedef __attribute__((ext_vector_type(8)))  __bf16 v8bf;
typedef __attribute__((ext_vector_type(8)))  float  v8f;
typedef __attribute__((ext_vector_type(4)))  float  v4f;

// ---------- workspace layout (bytes) ----------
#define OFF_S      ((size_t)0)            // f32  (B,LC,LQ)  33.5MB
#define OFF_S1     ((size_t)33554432)     // bf16 (B,LC,LQ)  16.8MB
#define OFF_E      ((size_t)50331648)     // bf16 (B,LQ,LC)  16.8MB (exp, unnormalized, transposed)
#define OFF_CM     ((size_t)67108864)     // bf16 (B,LC,D)   Cm = Ct*w_m  (A-operand)
#define OFF_CB     ((size_t)83886080)     // bf16 (B,D,LC)   C as-is      (B^T operand for T-GEMM)
#define OFF_QT     ((size_t)100663296)    // bf16 (B,LQ,D)   Q^T          (B^T operand for score GEMM)
#define OFF_QB     ((size_t)102760448)    // bf16 (B,D,LQ)   Q as-is      (B^T operand for A-GEMM)
#define OFF_TT     ((size_t)104857600)    // bf16 (B,D,LQ)   T^T          (B^T operand for Bt-GEMM)
#define OFF_SC     ((size_t)106954752)    // f32  (B,LC)
#define OFF_SQ     ((size_t)107216896)    // f32  (B,LQ)
#define OFF_CMAX   ((size_t)107249664)    // u32  (B,LQ) order-preserving keys
#define OFF_CSUM   ((size_t)107282432)    // f32  (B,LQ)

// order-preserving float<->uint map for atomicMax
__device__ __forceinline__ unsigned f2ord(float f) {
  unsigned u = __float_as_uint(f);
  return (u & 0x80000000u) ? ~u : (u | 0x80000000u);
}
__device__ __forceinline__ float ord2f(unsigned u) {
  return (u & 0x80000000u) ? __uint_as_float(u & 0x7fffffffu) : __uint_as_float(~u);
}

// A-fragment (16x32 bf16, M x K) from row-major tile, ld elements per row.
// ISA 7.12.2: lane(h,r): rows r, elems0..7 -> K = k0+h*8.., elems8..15 -> K = k0+16+h*8..
__device__ __forceinline__ v16bf frag_a(const __bf16* tile, int ld, int k0, int lane) {
  const int h = (lane >> 4) & 1, r = lane & 15;
  const __bf16* p = tile + r * ld + k0 + h * 8;
  v8bf lo = *(const v8bf*)(p);
  v8bf hi = *(const v8bf*)(p + 16);
  v16bf f;
#pragma unroll
  for (int i = 0; i < 8; ++i) { f[i] = lo[i]; f[8 + i] = hi[i]; }
  return f;
}
// B-fragment (32x16 bf16, K x N) loaded from the TRANSPOSED (N x K row-major) array bt.
// lane(h,n): column n0+n, elems0..7 -> K = k0+h*8.., elems8..15 -> K = k0+16+h*8..
__device__ __forceinline__ v16bf frag_b(const __bf16* bt, int ld, int n0, int k0, int lane) {
  const int h = (lane >> 4) & 1, n = lane & 15;
  const __bf16* p = bt + (size_t)(n0 + n) * ld + k0 + h * 8;
  v8bf lo = *(const v8bf*)(p);
  v8bf hi = *(const v8bf*)(p + 16);
  v16bf f;
#pragma unroll
  for (int i = 0; i < 8; ++i) { f[i] = lo[i]; f[8 + i] = hi[i]; }
  return f;
}
__device__ __forceinline__ v8f wmma_bf16(v16bf a, v16bf b, v8f c) {
  return __builtin_amdgcn_wmma_f32_16x16x32_bf16(false, a, false, b, (short)0, c, false, false);
}

// ---------------- K0: init colmax/colsum ----------------
__global__ void k0_init(unsigned* __restrict__ colmax, float* __restrict__ colsum) {
  int i = blockIdx.x * 256 + threadIdx.x;
  if (i < BB * LQq) { colmax[i] = 0u; colsum[i] = 0.f; }
}

// ---------------- K1q: Q-side prep ----------------
__global__ __launch_bounds__(256) void k1_prepq(const float* __restrict__ Q,
                                                const float* __restrict__ w,
                                                float* __restrict__ s_q,
                                                __bf16* __restrict__ Qt,
                                                __bf16* __restrict__ Qb) {
  int i = blockIdx.x * 256 + threadIdx.x;      // over B*LQ
  int b = i >> 7, q = i & 127;
  const float* qp = Q + (size_t)b * DD * LQq + q;
  float acc = 0.f;
  for (int d = 0; d < DD; ++d) {
    float v = qp[(size_t)d * LQq];
    acc += v * w[d];
    Qb[(size_t)b * DD * LQq + (size_t)d * LQq + q] = (__bf16)v;
    Qt[((size_t)b * LQq + q) * DD + d] = (__bf16)v;
  }
  s_q[i] = acc;
}

// ---------------- K1c: C-side prep ----------------
__global__ __launch_bounds__(256) void k1_prepc(const float* __restrict__ C,
                                                const float* __restrict__ w,
                                                float* __restrict__ s_c,
                                                __bf16* __restrict__ Cb,
                                                __bf16* __restrict__ Cm) {
  int i = blockIdx.x * 256 + threadIdx.x;      // over B*LC
  int b = i >> 10, c = i & 1023;
  float acc = 0.f;
  for (int d = 0; d < DD; ++d) {
    float v = C[((size_t)b * DD + d) * LCc + c];
    acc += v * w[DD + d];
    Cb[((size_t)b * DD + d) * LCc + c] = (__bf16)v;
    Cm[((size_t)b * LCc + c) * DD + d] = (__bf16)(v * w[2 * DD + d]);
  }
  s_c[i] = acc;
}

// ---------------- K2: scores + row softmax + colmax ----------------
__global__ __launch_bounds__(256) void k2_scores(const float* __restrict__ cmask,
                                                 const float* __restrict__ qmask,
                                                 const __bf16* __restrict__ Cm,
                                                 const __bf16* __restrict__ Qt,
                                                 const float* __restrict__ s_c,
                                                 const float* __restrict__ s_q,
                                                 float* __restrict__ S,
                                                 __bf16* __restrict__ S1,
                                                 unsigned* __restrict__ colmax) {
  __shared__ __align__(16) __bf16 qt_s[LQq * DD];   // 32 KB: B^T operand slice
  __shared__ float colred[8][LQq];                  // 4 KB
  const int b = blockIdx.y, c0 = blockIdx.x * 128;
  const int t = threadIdx.x, lane = t & 31, w = t >> 5;
  { // stage Qt[b] into LDS
    const uint4* src = (const uint4*)(Qt + (size_t)b * LQq * DD);
    uint4* dst = (uint4*)qt_s;
    for (int i = t; i < LQq * DD / 8; i += 256) dst[i] = src[i];
  }
  __syncthreads();

  const __bf16* atile = Cm + ((size_t)b * LCc + c0 + w * 16) * DD;
  v16bf a[4];
#pragma unroll
  for (int kt = 0; kt < 4; ++kt) a[kt] = frag_a(atile, DD, kt * 32, lane);

  v8f acc[8];
#pragma unroll
  for (int nt = 0; nt < 8; ++nt) {
    v8f cc = {0.f, 0.f, 0.f, 0.f, 0.f, 0.f, 0.f, 0.f};
#pragma unroll
    for (int kt = 0; kt < 4; ++kt)
      cc = wmma_bf16(a[kt], frag_b(qt_s, DD, nt * 16, kt * 32, lane), cc);
    acc[nt] = cc;
  }

  const int h = (lane >> 4) & 1, n = lane & 15;
  float scv[8], cpen[8], sqv[8], qpen[8];
#pragma unroll
  for (int j = 0; j < 8; ++j) {
    int c = c0 + w * 16 + h * 8 + j;
    scv[j] = s_c[b * LCc + c];
    cpen[j] = (1.f - cmask[b * LCc + c]) * NEGV;
  }
#pragma unroll
  for (int nt = 0; nt < 8; ++nt) {
    int q = nt * 16 + n;
    sqv[nt] = s_q[b * LQq + q];
    qpen[nt] = (1.f - qmask[b * LQq + q]) * NEGV;
  }
  // raw scores S = s_m + s_c + s_q ; store
#pragma unroll
  for (int nt = 0; nt < 8; ++nt)
#pragma unroll
    for (int j = 0; j < 8; ++j) {
      acc[nt][j] += scv[j] + sqv[nt];
      int c = c0 + w * 16 + h * 8 + j;
      S[((size_t)b * LCc + c) * LQq + nt * 16 + n] = acc[nt][j];
    }
  // column partial max (with cmask penalty), reduce h-halves via lane^16
  float pmax[8];
#pragma unroll
  for (int nt = 0; nt < 8; ++nt) {
    float m = -3.0e38f;
#pragma unroll
    for (int j = 0; j < 8; ++j) m = fmaxf(m, acc[nt][j] + cpen[j]);
    m = fmaxf(m, __shfl_xor(m, 16, 32));
    pmax[nt] = m;
  }
  // row softmax (with qmask penalty) over q: reduce across 16-lane group
  float rmax[8], rsum[8];
#pragma unroll
  for (int j = 0; j < 8; ++j) {
    float m = -3.0e38f;
#pragma unroll
    for (int nt = 0; nt < 8; ++nt) m = fmaxf(m, acc[nt][j] + qpen[nt]);
    m = fmaxf(m, __shfl_xor(m, 1, 32));
    m = fmaxf(m, __shfl_xor(m, 2, 32));
    m = fmaxf(m, __shfl_xor(m, 4, 32));
    m = fmaxf(m, __shfl_xor(m, 8, 32));
    rmax[j] = m;
  }
#pragma unroll
  for (int j = 0; j < 8; ++j) {
    float s = 0.f;
#pragma unroll
    for (int nt = 0; nt < 8; ++nt) {
      float e = __expf(acc[nt][j] + qpen[nt] - rmax[j]);
      acc[nt][j] = e;               // reuse accumulators for exp values
      s += e;
    }
    s += __shfl_xor(s, 1, 32);
    s += __shfl_xor(s, 2, 32);
    s += __shfl_xor(s, 4, 32);
    s += __shfl_xor(s, 8, 32);
    rsum[j] = s;
  }
#pragma unroll
  for (int nt = 0; nt < 8; ++nt)
#pragma unroll
    for (int j = 0; j < 8; ++j) {
      int c = c0 + w * 16 + h * 8 + j;
      S1[((size_t)b * LCc + c) * LQq + nt * 16 + n] = (__bf16)(acc[nt][j] / rsum[j]);
    }
  // cross-wave column-max reduce, then one global atomic per column per WG
  if (h == 0) {
#pragma unroll
    for (int nt = 0; nt < 8; ++nt) colred[w][nt * 16 + n] = pmax[nt];
  }
  __syncthreads();
  if (t < LQq) {
    float m = colred[0][t];
#pragma unroll
    for (int ww = 1; ww < 8; ++ww) m = fmaxf(m, colred[ww][t]);
    atomicMax(&colmax[b * LQq + t], f2ord(m));
  }
}

// ---------------- K3: exp(S-colmax), transpose to (q,c), colsum ----------------
__global__ __launch_bounds__(256) void k3_expt(const float* __restrict__ S,
                                               const float* __restrict__ cmask,
                                               const unsigned* __restrict__ colmax,
                                               float* __restrict__ colsum,
                                               __bf16* __restrict__ E) {
  __shared__ __align__(16) __bf16 et[128 * 136];    // pitch 136 elems = 272B (16B aligned)
  __shared__ float psum[256];
  const int b = blockIdx.y, c0 = blockIdx.x * 128, t = threadIdx.x;
  const int q = t & 127;
  const float mx = ord2f(colmax[b * LQq + q]);
  float ls = 0.f;
  for (int k = 0; k < 64; ++k) {
    int cl = (t >> 7) + 2 * k;
    int c = c0 + cl;
    float v = S[((size_t)b * LCc + c) * LQq + q] + (1.f - cmask[b * LCc + c]) * NEGV;
    float e = __expf(v - mx);
    ls += e;
    et[q * 136 + cl] = (__bf16)e;
  }
  psum[t] = ls;
  __syncthreads();
  if (t < LQq) atomicAdd(&colsum[b * LQq + t], psum[t] + psum[t + 128]);
  // write transposed tile: row q, 64 bf16 per thread, contiguous
  const int qq = t >> 1, half = t & 1;
  uint4* dst = (uint4*)(E + ((size_t)b * LQq + qq) * LCc + c0 + half * 64);
  const uint4* src = (const uint4*)(et + qq * 136 + half * 64);
#pragma unroll
  for (int i = 0; i < 8; ++i) dst[i] = src[i];
}

// ---------------- K4: T = (E @ C^T)/colsum, store T^T ----------------
__global__ __launch_bounds__(256) void k4_tgemm(const __bf16* __restrict__ E,
                                                const __bf16* __restrict__ Cb,
                                                const float* __restrict__ colsum,
                                                __bf16* __restrict__ Tt) {
  const int b = blockIdx.x, t = threadIdx.x, lane = t & 31, w = t >> 5;
  const __bf16* atile = E + ((size_t)b * LQq + w * 16) * LCc;
  const __bf16* btile = Cb + (size_t)b * DD * LCc;
  v8f acc[8];
#pragma unroll
  for (int nt = 0; nt < 8; ++nt) acc[nt] = (v8f){0.f, 0.f, 0.f, 0.f, 0.f, 0.f, 0.f, 0.f};
  for (int c0 = 0; c0 < LCc; c0 += 32) {
    __builtin_prefetch(atile + (lane & 15) * LCc + c0 + 256, 0, 1);
    v16bf a = frag_a(atile, LCc, c0, lane);
#pragma unroll
    for (int nt = 0; nt < 8; ++nt)
      acc[nt] = wmma_bf16(a, frag_b(btile, LCc, nt * 16, c0, lane), acc[nt]);
  }
  const int h = (lane >> 4) & 1, n = lane & 15;
  float inv[8];
#pragma unroll
  for (int j = 0; j < 8; ++j) inv[j] = 1.f / colsum[b * LQq + w * 16 + h * 8 + j];
#pragma unroll
  for (int nt = 0; nt < 8; ++nt)
#pragma unroll
    for (int j = 0; j < 8; ++j) {
      int d = nt * 16 + n, q = w * 16 + h * 8 + j;
      Tt[((size_t)b * DD + d) * LQq + q] = (__bf16)(acc[nt][j] * inv[j]);
    }
}

// ---------------- K5: A = S1@Qt, Bt = S1@T, fused transposed epilogue ----------------
__global__ __launch_bounds__(256) void k5_final(const float* __restrict__ C,
                                                const __bf16* __restrict__ S1,
                                                const __bf16* __restrict__ Qb,
                                                const __bf16* __restrict__ Tt,
                                                float* __restrict__ out) {
  __shared__ __align__(16) __bf16 qb_s[DD * LQq];   // 32 KB
  __shared__ __align__(16) __bf16 tt_s[DD * LQq];   // 32 KB
  const int b = blockIdx.y, c0 = blockIdx.x * 128;
  const int t = threadIdx.x, lane = t & 31, w = t >> 5;
  {
    const uint4* s0 = (const uint4*)(Qb + (size_t)b * DD * LQq);
    const uint4* s1 = (const uint4*)(Tt + (size_t)b * DD * LQq);
    uint4* d0 = (uint4*)qb_s;
    uint4* d1 = (uint4*)tt_s;
    for (int i = t; i < DD * LQq / 8; i += 256) { d0[i] = s0[i]; d1[i] = s1[i]; }
  }
  __syncthreads();
  const __bf16* atile = S1 + ((size_t)b * LCc + c0 + w * 16) * LQq;
  v16bf a[4];
#pragma unroll
  for (int kt = 0; kt < 4; ++kt) a[kt] = frag_a(atile, LQq, kt * 32, lane);
  v8f accA[8], accB[8];
#pragma unroll
  for (int nt = 0; nt < 8; ++nt) {
    v8f ca = {0.f, 0.f, 0.f, 0.f, 0.f, 0.f, 0.f, 0.f};
    v8f cb = {0.f, 0.f, 0.f, 0.f, 0.f, 0.f, 0.f, 0.f};
#pragma unroll
    for (int kt = 0; kt < 4; ++kt) {
      ca = wmma_bf16(a[kt], frag_b(qb_s, LQq, nt * 16, kt * 32, lane), ca);
      cb = wmma_bf16(a[kt], frag_b(tt_s, LQq, nt * 16, kt * 32, lane), cb);
    }
    accA[nt] = ca;
    accB[nt] = cb;
  }
  const int h = (lane >> 4) & 1, n = lane & 15;
  const int crow = c0 + w * 16 + h * 8;            // 8 consecutive c per lane
  const size_t ob = (size_t)b * 4 * DD * LCc;
#pragma unroll
  for (int nt = 0; nt < 8; ++nt) {
    int d = nt * 16 + n;
    const float* cp = C + ((size_t)b * DD + d) * LCc + crow;
    v4f cv0 = *(const v4f*)cp, cv1 = *(const v4f*)(cp + 4);
    v4f a0 = {accA[nt][0], accA[nt][1], accA[nt][2], accA[nt][3]};
    v4f a1 = {accA[nt][4], accA[nt][5], accA[nt][6], accA[nt][7]};
    v4f b0 = {accB[nt][0], accB[nt][1], accB[nt][2], accB[nt][3]};
    v4f b1 = {accB[nt][4], accB[nt][5], accB[nt][6], accB[nt][7]};
    float* o1 = out + ob + (size_t)(DD + d) * LCc + crow;
    float* o2 = out + ob + (size_t)(2 * DD + d) * LCc + crow;
    float* o3 = out + ob + (size_t)(3 * DD + d) * LCc + crow;
    *(v4f*)o1 = a0;        *(v4f*)(o1 + 4) = a1;
    *(v4f*)o2 = cv0 * a0;  *(v4f*)(o2 + 4) = cv1 * a1;
    *(v4f*)o3 = cv0 * b0;  *(v4f*)(o3 + 4) = cv1 * b1;
  }
}

// ---------------- K6: out[b,0:D,:] = C ----------------
__global__ void k6_copy(const float* __restrict__ C, float* __restrict__ out) {
  size_t i = (size_t)blockIdx.x * 256 + threadIdx.x;  // over B*D*LC/4
  const size_t per = (size_t)DD * LCc / 4;
  size_t b = i / per, r = i % per;
  ((float4*)out)[b * ((size_t)4 * DD * LCc / 4) + r] = ((const float4*)C)[b * per + r];
}

extern "C" void kernel_launch(void* const* d_in, const int* in_sizes, int n_in,
                              void* d_out, int out_size, void* d_ws, size_t ws_size,
                              hipStream_t stream) {
  const float* C = (const float*)d_in[0];
  const float* Q = (const float*)d_in[1];
  const float* cmask = (const float*)d_in[2];
  const float* qmask = (const float*)d_in[3];
  const float* w = (const float*)d_in[4];
  float* out = (float*)d_out;
  char* ws = (char*)d_ws;

  float*    S      = (float*)(ws + OFF_S);
  __bf16*   S1     = (__bf16*)(ws + OFF_S1);
  __bf16*   E      = (__bf16*)(ws + OFF_E);
  __bf16*   Cm     = (__bf16*)(ws + OFF_CM);
  __bf16*   Cb     = (__bf16*)(ws + OFF_CB);
  __bf16*   Qt     = (__bf16*)(ws + OFF_QT);
  __bf16*   Qb     = (__bf16*)(ws + OFF_QB);
  __bf16*   Tt     = (__bf16*)(ws + OFF_TT);
  float*    s_c    = (float*)(ws + OFF_SC);
  float*    s_q    = (float*)(ws + OFF_SQ);
  unsigned* colmax = (unsigned*)(ws + OFF_CMAX);
  float*    colsum = (float*)(ws + OFF_CSUM);

  k0_init<<<32, 256, 0, stream>>>(colmax, colsum);
  k1_prepq<<<32, 256, 0, stream>>>(Q, w, s_q, Qt, Qb);
  k1_prepc<<<256, 256, 0, stream>>>(C, w, s_c, Cb, Cm);
  k2_scores<<<dim3(8, 64), 256, 0, stream>>>(cmask, qmask, Cm, Qt, s_c, s_q, S, S1, colmax);
  k3_expt<<<dim3(8, 64), 256, 0, stream>>>(S, cmask, colmax, colsum, E);
  k4_tgemm<<<64, 256, 0, stream>>>(E, Cb, colsum, Tt);
  k5_final<<<dim3(8, 64), 256, 0, stream>>>(C, S1, Qb, Tt, out);
  k6_copy<<<8192, 256, 0, stream>>>(C, out);
}
